// QKVAttentionLegacy_14955076125229
// MI455X (gfx1250) — matrix operand order repaired and checked by reading, not compiled, not run on hardware
//
#include <hip/hip_runtime.h>

typedef __attribute__((ext_vector_type(16))) _Float16 v16h;
typedef __attribute__((ext_vector_type(8)))  _Float16 v8h;
typedef __attribute__((ext_vector_type(2)))  _Float16 h2;
typedef __attribute__((ext_vector_type(8)))  float    v8f;
typedef __attribute__((ext_vector_type(4)))  float    f32x4;

#define T_LEN 2048
#define CH    64
#define TS    32              // keys per tile
#define NT    (T_LEN / TS)    // 64 tiles
#define WAVES 8
#define BLOCK_ROWS 128

// LDS pitches in halves; multiples of 8 halves (16B) for aligned v8h access
#define KPITCH 72     // kS[s][c] rows: 144B each
#define VPITCH 40     // vS[c][p] rows: 80B each (p = interleaved s position)
#define PPITCH 40     // pS[w][t][p] rows: 80B each

static __device__ __forceinline__ v16h join16(v8h lo, v8h hi) {
    v16h r;
#pragma unroll
    for (int j = 0; j < 8; ++j) { r[j] = lo[j]; r[j + 8] = hi[j]; }
    return r;
}

__global__ __launch_bounds__(256)
void attn_flash_wmma(const float* __restrict__ qkv,
                     const float* __restrict__ rescale,
                     float* __restrict__ out)
{
    // double-buffered K (transposed to [s][c]) and V ([c][interleaved-s]) tiles
    __shared__ __align__(32) _Float16 kS[2][TS][KPITCH];
    __shared__ __align__(32) _Float16 vS[2][CH][VPITCH];
    __shared__ __align__(32) _Float16 pS[WAVES][16][PPITCH];   // per-wave P round-trip

    const int bid   = blockIdx.x;
    const int b     = bid >> 4;          // 0..31  (batch*head)
    const int ttile = bid & 15;          // query tile within T
    const int n = b >> 3, h = b & 7;

    const size_t headBase = ((size_t)n * 1536 + (size_t)h * 192) * T_LEN;
    const float* qp = qkv + headBase;
    const float* kp = qkv + headBase + (size_t)64  * T_LEN;
    const float* vp = qkv + headBase + (size_t)128 * T_LEN;
    float* op = out + ((size_t)n * 512 + (size_t)h * 64) * T_LEN;

    const int tid    = threadIdx.x;
    const int wave   = tid >> 5;
    const int lane   = tid & 31;
    const int hilane = lane >> 4;        // 0: lanes 0-15, 1: lanes 16-31
    const int l16    = lane & 15;

    const float qscale = 0.35355339059327373f;   // 64^-0.25
    // fold rescale AND log2(e) into Q so P = exp2(S) with no per-element mul
    const float qmul   = qscale * rescale[0] * 1.4426950408889634f;

    // staging coordinates: 256 threads cover a 64c x 32s tile;
    // each thread owns s' in {sg..sg+3} U {sg+16..sg+19}
    const int cstg = tid >> 2;
    const int sg   = (tid & 3) * 4;
    const float* kstg = kp + (size_t)cstg * T_LEN + sg;
    const float* vstg = vp + (size_t)cstg * T_LEN + sg;

    // ---- load Q A-operands (rows t, cols c) in WMMA A layout ----
    const int trow = ttile * BLOCK_ROWS + wave * 16 + l16;   // A-layout row = lane%16
    v16h qa0, qa1;
#pragma unroll
    for (int j = 0; j < 16; ++j) {
        int c = j + hilane * 16;                             // K index, contiguous order k = 16*hilane + j (same as B)
        qa0[j] = (_Float16)(qp[(size_t)c        * T_LEN + trow] * qmul);
        qa1[j] = (_Float16)(qp[(size_t)(c + 32) * T_LEN + trow] * qmul);
    }

    v16h ones;                                   // B = all-ones for row-sum WMMA
#pragma unroll
    for (int j = 0; j < 16; ++j) ones[j] = (_Float16)1.0f;

    v8f acc0 = {}, acc1 = {}, acc2 = {}, acc3 = {};  // O tile: 16 t x 64 c
    v8f accL = {};                                   // row sums of P (softmax denom)

    // ---- tile staging: f32 global -> f16 LDS ----
    // kS natural order (transposed scatter); vS interleaved: pos p=2*(s'%16)+s'/16
    auto stage = [&](int buf, int s0) {
        f32x4 k0 = *(const f32x4*)(kstg + s0);
        f32x4 k1 = *(const f32x4*)(kstg + s0 + 16);
        f32x4 v0 = *(const f32x4*)(vstg + s0);
        f32x4 v1 = *(const f32x4*)(vstg + s0 + 16);
        v8h pv;
#pragma unroll
        for (int u = 0; u < 4; ++u) {
            kS[buf][sg + u][cstg]      = (_Float16)(k0[u] * qscale);
            kS[buf][sg + 16 + u][cstg] = (_Float16)(k1[u] * qscale);
            pv[2 * u]     = (_Float16)v0[u];     // pos 2*(sg+u)   <- s'=sg+u
            pv[2 * u + 1] = (_Float16)v1[u];     // pos 2*(sg+u)+1 <- s'=sg+16+u
        }
        *(v8h*)&vS[buf][cstg][2 * sg] = pv;      // one ds_store_b128
    };

    stage(0, 0);
    __syncthreads();

    for (int it = 0; it < NT; ++it) {
        const int cur = it & 1;
        if (it + 1 < NT) {
            stage(cur ^ 1, (it + 1) * TS);            // overlap with compute
            if (it + 2 < NT) {                        // global_prefetch tile+2
                __builtin_prefetch(kstg + (it + 2) * TS, 0, 3);
                __builtin_prefetch(vstg + (it + 2) * TS, 0, 3);
            }
        }

        // ---- load all four QK B operands, then issue the four WMMAs ----
        const _Float16* kr0 = &kS[cur][l16][hilane * 16];        // S0: rows 0..15
        const _Float16* kr1 = &kS[cur][16 + l16][hilane * 16];   // S1: rows 16..31
        v16h b00 = join16(*(const v8h*)(kr0),      *(const v8h*)(kr0 + 8));   // c 0..31
        v16h b01 = join16(*(const v8h*)(kr0 + 32), *(const v8h*)(kr0 + 40));  // c 32..63
        v16h b10 = join16(*(const v8h*)(kr1),      *(const v8h*)(kr1 + 8));
        v16h b11 = join16(*(const v8h*)(kr1 + 32), *(const v8h*)(kr1 + 40));

        v8f S0 = {}, S1 = {};
        S0 = __builtin_amdgcn_wmma_f32_16x16x32_f16(false, qa0, false, b00,
                                                    (short)0, S0, false, false);
        S1 = __builtin_amdgcn_wmma_f32_16x16x32_f16(false, qa0, false, b10,
                                                    (short)0, S1, false, false);
        S0 = __builtin_amdgcn_wmma_f32_16x16x32_f16(false, qa1, false, b01,
                                                    (short)0, S0, false, false);
        S1 = __builtin_amdgcn_wmma_f32_16x16x32_f16(false, qa1, false, b11,
                                                    (short)0, S1, false, false);

        // ---- P = exp2(S)  (log2e pre-folded into Q; scores are O(1)) ----
        // ---- packed C-layout -> A-layout round-trip: one b32 store per row ----
#pragma unroll
        for (int i = 0; i < 8; ++i) {
            h2 ph;
            ph[0] = (_Float16)__builtin_amdgcn_exp2f(S0[i]);  // pos 2*l16   (s'=l16)
            ph[1] = (_Float16)__builtin_amdgcn_exp2f(S1[i]);  // pos 2*l16+1 (s'=16+l16)
            *(h2*)&pS[wave][hilane * 8 + i][2 * l16] = ph;
        }
        const _Float16* pr = &pS[wave][l16][0];
        v16h pa = join16(*(const v8h*)(pr + hilane * 16),
                         *(const v8h*)(pr + hilane * 16 + 8));

        // ---- row sums of P on the matrix pipe: accL += P x ones ----
        accL = __builtin_amdgcn_wmma_f32_16x16x32_f16(false, pa, false, ones,
                                                      (short)0, accL, false, false);

        // ---- O += P x V^T (4 column chunks of 16 channels) ----
        {
            const _Float16* vr0 = &vS[cur][0  + l16][hilane * 16];
            const _Float16* vr1 = &vS[cur][16 + l16][hilane * 16];
            const _Float16* vr2 = &vS[cur][32 + l16][hilane * 16];
            const _Float16* vr3 = &vS[cur][48 + l16][hilane * 16];
            v16h vb0 = join16(*(const v8h*)(vr0), *(const v8h*)(vr0 + 8));
            v16h vb1 = join16(*(const v8h*)(vr1), *(const v8h*)(vr1 + 8));
            v16h vb2 = join16(*(const v8h*)(vr2), *(const v8h*)(vr2 + 8));
            v16h vb3 = join16(*(const v8h*)(vr3), *(const v8h*)(vr3 + 8));
            acc0 = __builtin_amdgcn_wmma_f32_16x16x32_f16(false, pa, false, vb0,
                                                          (short)0, acc0, false, false);
            acc1 = __builtin_amdgcn_wmma_f32_16x16x32_f16(false, pa, false, vb1,
                                                          (short)0, acc1, false, false);
            acc2 = __builtin_amdgcn_wmma_f32_16x16x32_f16(false, pa, false, vb2,
                                                          (short)0, acc2, false, false);
            acc3 = __builtin_amdgcn_wmma_f32_16x16x32_f16(false, pa, false, vb3,
                                                          (short)0, acc3, false, false);
        }
        __syncthreads();   // staged next buffer visible; current buffer reusable
    }

    // ---- epilogue: normalize rows, packed b128 stores (8 consecutive t) ----
    float inv[8];
#pragma unroll
    for (int i = 0; i < 8; ++i) inv[i] = 1.0f / accL[i];
    const int tbase = ttile * BLOCK_ROWS + wave * 16 + hilane * 8;

    const v8f* accs[4] = { &acc0, &acc1, &acc2, &acc3 };
#pragma unroll
    for (int j = 0; j < 4; ++j) {
        const v8f& a = *accs[j];
        f32x4 o0, o1;
#pragma unroll
        for (int u = 0; u < 4; ++u) { o0[u] = a[u] * inv[u]; o1[u] = a[4 + u] * inv[4 + u]; }
        float* dst = op + (size_t)(j * 16 + l16) * T_LEN + tbase;
        *(f32x4*)(dst)     = o0;
        *(f32x4*)(dst + 4) = o1;
    }
}

extern "C" void kernel_launch(void* const* d_in, const int* in_sizes, int n_in,
                              void* d_out, int out_size, void* d_ws, size_t ws_size,
                              hipStream_t stream) {
    const float* qkv     = (const float*)d_in[0];
    const float* rescale = (const float*)d_in[1];
    float* out           = (float*)d_out;
    // 32 (batch*head) slices x 16 query tiles of 128 rows
    dim3 grid(512), block(256);
    attn_flash_wmma<<<grid, block, 0, stream>>>(qkv, rescale, out);
}
